// FeatureAlign_88862873354876
// MI455X (gfx1250) — matrix operand order repaired and checked by reading, not compiled, not run on hardware
//
#include <hip/hip_runtime.h>
#include <math.h>

#define BB   4
#define CC   128
#define HH   128
#define WW   128
#define HW_  (HH*WW)
#define OC2  216
#define OC2P 224
#define K1   256
#define K2   1152
#define LDSD 40   // LDS row stride (bf16): 80B rows -> 16B-aligned 8-elem stores, staggered banks

typedef __attribute__((ext_vector_type(16))) __bf16 v16bf;
typedef __attribute__((ext_vector_type(8)))  __bf16 v8bf;
typedef __attribute__((ext_vector_type(8)))  float  v8f;

__device__ __forceinline__ float  bf2f(__bf16 v){ return (float)v; }
__device__ __forceinline__ __bf16 f2bf(float v){ return (__bf16)v; }

__device__ __forceinline__ v8f wmma_bf16(v16bf a, v16bf b, v8f c){
  // emits v_wmma_f32_16x16x32_bf16
  return __builtin_amdgcn_wmma_f32_16x16x32_bf16(false, a, false, b, (short)0, c, false, false);
}

// A fragment from row-major bf16 [M x ldk]; rows row_base + (lane&15)
// lane<16: K = kb+0..15 ; lane>=16: K = kb+16..31
__device__ __forceinline__ v16bf load_a(const __bf16* Wmat, int row_base, int ldk, int kb, int lane){
  int row   = row_base + (lane & 15);
  int khalf = (lane >> 4) * 16;
  const __bf16* p = Wmat + row * ldk + kb + khalf;
  v16bf a;
#pragma unroll
  for (int i = 0; i < 8; ++i){ a[i] = p[i]; a[8+i] = p[8+i]; }
  return a;
}

// B fragment from LDS tile stored [n][LDSD] (k contiguous per n-row).
// lane: col N = nbase + (lane&15), K = (lane>>4)*16 + 0..15 (contiguous 32B read)
__device__ __forceinline__ v16bf load_b(const __bf16* lds, int nbase, int lane){
  const __bf16* p = lds + (nbase + (lane & 15)) * LDSD + (lane >> 4) * 16;
  v16bf b;
#pragma unroll
  for (int i = 0; i < 16; ++i) b[i] = p[i];
  return b;
}

// ---------------- weight packing (fp32 -> packed bf16 GEMM layouts) ----------------
__global__ __launch_bounds__(256) void prep_weights(const float* __restrict__ W_off,
                                                    const float* __restrict__ W_om,
                                                    const float* __restrict__ W_dcn,
                                                    unsigned short* Wo_, unsigned short* Wm_, unsigned short* Wd_){
  __bf16* Wo = (__bf16*)Wo_;
  __bf16* Wm = (__bf16*)Wm_;
  __bf16* Wd = (__bf16*)Wd_;
  int id = blockIdx.x * 256 + threadIdx.x;
  if (id < CC * K1) Wo[id] = f2bf(W_off[id]);             // [oc][ic] row-major
  if (id < OC2P * K2){                                    // k = kappa*128 + ic
    int oc = id / K2, k = id % K2;
    int kap = k / CC, ic = k % CC;
    Wm[id] = (oc < OC2) ? f2bf(W_om[(oc * CC + ic) * 9 + kap]) : f2bf(0.f);
  }
  if (id < CC * K2){                                      // k = (g*9+kk)*16 + c
    int o = id / K2, k = id % K2;
    int p = k / 16, c = k % 16;
    int g = p / 9, kk = p % 9;
    Wd[id] = f2bf(W_dcn[(o * CC + g * 16 + c) * 9 + kk]);
  }
}

// ---------------- bilinear 2x upsample (half-pixel), fp32 -> bf16 ----------------
__global__ __launch_bounds__(256) void upsample2x(const float* __restrict__ fs, unsigned short* fu_){
  __bf16* fu = (__bf16*)fu_;
  int id = blockIdx.x * 256 + threadIdx.x;   // covers BB*CC*HW_ exactly
  int x  = id & (WW - 1);
  int y  = (id >> 7) & (HH - 1);
  int bc = id >> 14;
  const float* src = fs + bc * 64 * 64;
  float sy = 0.5f * (float)y - 0.25f;
  float sx = 0.5f * (float)x - 0.25f;
  float y0f = floorf(sy), x0f = floorf(sx);
  float fy = sy - y0f, fx = sx - x0f;
  int y0 = (int)y0f, x0 = (int)x0f;
  int y0c = min(max(y0, 0), 63), y1c = min(max(y0 + 1, 0), 63);
  int x0c = min(max(x0, 0), 63), x1c = min(max(x0 + 1, 0), 63);
  float v = (1.f - fy) * ((1.f - fx) * src[y0c * 64 + x0c] + fx * src[y0c * 64 + x1c])
          +        fy  * ((1.f - fx) * src[y1c * 64 + x0c] + fx * src[y1c * 64 + x1c]);
  fu[id] = f2bf(v);
}

// ---------------- GEMM1: 1x1 conv, out[128 x N] = Wo[128x256] * X[256 x N] ----------------
// X[ic,n] = ic<128 ? feat_l : 2*feat_up. Block: 128 thr (4 waves), N-tile 32, full M=128.
__global__ __launch_bounds__(128) void gemm1(const float* __restrict__ feat_l,
                                             const unsigned short* fu_,
                                             const unsigned short* Wo_,
                                             unsigned short* ofeat_){
  const __bf16* fu = (const __bf16*)fu_;
  const __bf16* Wo = (const __bf16*)Wo_;
  __bf16* ofeat = (__bf16*)ofeat_;
  __shared__ __align__(16) __bf16 lds[32 * LDSD];
  int t = threadIdx.x, lane = t & 31, wid = t >> 5;
  int P = blockIdx.x * 32;
  int b = P >> 14, hw0 = P & (HW_ - 1);
  v8f acc[2][2] = {};
  for (int kb = 0; kb < K1; kb += 32){
    int n = t & 31, kg = t >> 5;
    v8bf sv;
    if (kb < CC){               // uniform scalar branch: feat_l half of K
      const float* src = feat_l + (b * CC + kb + kg * 8) * HW_ + hw0 + n;
#pragma unroll
      for (int i = 0; i < 8; ++i) sv[i] = f2bf(src[i * HW_]);
    } else {                    // feat_up half of K, scaled by 2
      const __bf16* src = fu + (b * CC + (kb - CC) + kg * 8) * HW_ + hw0 + n;
#pragma unroll
      for (int i = 0; i < 8; ++i) sv[i] = f2bf(2.f * bf2f(src[i * HW_]));
    }
    *(v8bf*)(lds + n * LDSD + kg * 8) = sv;
    __syncthreads();
    v16bf bf0 = load_b(lds, 0, lane);
    v16bf bf1 = load_b(lds, 16, lane);
    v16bf a0  = load_a(Wo, wid * 32,      K1, kb, lane);
    v16bf a1  = load_a(Wo, wid * 32 + 16, K1, kb, lane);
    acc[0][0] = wmma_bf16(a0, bf0, acc[0][0]);
    acc[0][1] = wmma_bf16(a0, bf1, acc[0][1]);
    acc[1][0] = wmma_bf16(a1, bf0, acc[1][0]);
    acc[1][1] = wmma_bf16(a1, bf1, acc[1][1]);
    __syncthreads();
  }
  int col = lane & 15, hi = lane >> 4;
#pragma unroll
  for (int mt = 0; mt < 2; ++mt)
#pragma unroll
    for (int nt = 0; nt < 2; ++nt)
#pragma unroll
      for (int r = 0; r < 8; ++r){
        int oc = wid * 32 + mt * 16 + r + 8 * hi;
        int hw = hw0 + nt * 16 + col;
        ofeat[(b * CC + oc) * HW_ + hw] = f2bf(acc[mt][nt][r]);
      }
}

// ---------------- GEMM2: 3x3 conv, implicit im2col, M=224(216), K=1152 ----------------
// Block: 224 thr (7 waves). Epilogue: +bias; sigmoid on mask channels (oc>=144).
__global__ __launch_bounds__(224) void gemm2(const unsigned short* ofeat_,
                                             const unsigned short* Wm_,
                                             const float* __restrict__ b_om,
                                             float* __restrict__ om){
  const __bf16* ofeat = (const __bf16*)ofeat_;
  const __bf16* Wm = (const __bf16*)Wm_;
  __shared__ __align__(16) __bf16 lds[32 * LDSD];
  int t = threadIdx.x, lane = t & 31, wid = t >> 5;
  int P = blockIdx.x * 32;
  int b = P >> 14, hw0 = P & (HW_ - 1);
  int h = hw0 >> 7, w0 = hw0 & (WW - 1);
  v8f acc[2][2] = {};
  for (int kt = 0; kt < K2 / 32; ++kt){
    int kap = kt >> 2, ic0 = (kt & 3) * 32;
    int dy = kap / 3 - 1, dx = kap % 3 - 1;
    if (t < 128){
      int n = t & 31, kg = t >> 5;
      int y = h + dy, x = w0 + n + dx;
      bool ok = ((unsigned)y < (unsigned)HH) && ((unsigned)x < (unsigned)WW);
      int yc = min(max(y, 0), HH - 1), xc = min(max(x, 0), WW - 1);
      const __bf16* src = ofeat + (b * CC + ic0 + kg * 8) * HW_ + yc * WW + xc;
      v8bf sv;
#pragma unroll
      for (int i = 0; i < 8; ++i){
        float v = bf2f(src[i * HW_]);           // unconditional clamped load
        sv[i] = f2bf(ok ? v : 0.f);             // v_cndmask, no exec branching
      }
      *(v8bf*)(lds + n * LDSD + kg * 8) = sv;
    }
    __syncthreads();
    v16bf bf0 = load_b(lds, 0, lane);
    v16bf bf1 = load_b(lds, 16, lane);
    v16bf a0  = load_a(Wm, wid * 32,      K2, kt * 32, lane);
    v16bf a1  = load_a(Wm, wid * 32 + 16, K2, kt * 32, lane);
    acc[0][0] = wmma_bf16(a0, bf0, acc[0][0]);
    acc[0][1] = wmma_bf16(a0, bf1, acc[0][1]);
    acc[1][0] = wmma_bf16(a1, bf0, acc[1][0]);
    acc[1][1] = wmma_bf16(a1, bf1, acc[1][1]);
    __syncthreads();
  }
  int col = lane & 15, hi = lane >> 4;
#pragma unroll
  for (int mt = 0; mt < 2; ++mt)
#pragma unroll
    for (int nt = 0; nt < 2; ++nt)
#pragma unroll
      for (int r = 0; r < 8; ++r){
        int oc = wid * 32 + mt * 16 + r + 8 * hi;
        if (oc < OC2){
          int hw = hw0 + nt * 16 + col;
          float v = acc[mt][nt][r] + b_om[oc];
          if (oc >= 144) v = 1.f / (1.f + expf(-v));   // mask -> sigmoid
          om[(b * OC2 + oc) * HW_ + hw] = v;
        }
      }
}

// ---------------- DCNv2: gather/bilinear into LDS val-tile, WMMA over K=1152 ----------------
// k = (g*9+kk)*16 + c. Epilogue: relu(acc+bias) + feat_l -> out.
__global__ __launch_bounds__(128) void dcn_wmma(const unsigned short* fu_,
                                                const float* __restrict__ om,
                                                const unsigned short* Wd_,
                                                const float* __restrict__ b_dcn,
                                                const float* __restrict__ feat_l,
                                                float* __restrict__ out){
  const __bf16* fu = (const __bf16*)fu_;
  const __bf16* Wd = (const __bf16*)Wd_;
  __shared__ __align__(16) __bf16 lds[32 * LDSD];
  int t = threadIdx.x, lane = t & 31, wid = t >> 5;
  int P = blockIdx.x * 32;
  int b = P >> 14, hw0 = P & (HW_ - 1);
  int h = hw0 >> 7, w0 = hw0 & (WW - 1);
  v8f acc[2][2] = {};
  for (int kt = 0; kt < K2 / 32; ++kt){
    // stage val[k][n] for two (g,kk) taps: pair p = 2*kt + pl
    {
      int combo = t >> 1;              // 64 combos = 2 taps x 32 pixels
      int pl = combo >> 5, n = combo & 31, c0 = (t & 1) * 8;
      int p  = kt * 2 + pl;
      int g  = p / 9, kk = p % 9;
      int ky = kk / 3, kx = kk % 3;
      int hw = hw0 + n;
      const float* omb = om + (size_t)b * OC2 * HW_;
      float offy = omb[(g * 18 + kk * 2    ) * HW_ + hw];
      float offx = omb[(g * 18 + kk * 2 + 1) * HW_ + hw];
      float msk  = omb[(144 + g * 9 + kk   ) * HW_ + hw];
      float py = (float)h        - 1.f + (float)ky + offy;
      float px = (float)(w0 + n) - 1.f + (float)kx + offx;
      float y0f = floorf(py), x0f = floorf(px);
      int y0 = (int)y0f, x0 = (int)x0f;
      float ly = py - y0f, lx = px - x0f;
      int y1 = y0 + 1, x1 = x0 + 1;
      bool vy0 = (unsigned)y0 < (unsigned)HH, vy1 = (unsigned)y1 < (unsigned)HH;
      bool vx0 = (unsigned)x0 < (unsigned)WW, vx1 = (unsigned)x1 < (unsigned)WW;
      // fold validity into bilinear weights (and mask) -> all 4 gathers unconditional
      float w00 = (vy0 && vx0) ? (1.f - ly) * (1.f - lx) * msk : 0.f;
      float w01 = (vy0 && vx1) ? (1.f - ly) * lx         * msk : 0.f;
      float w10 = (vy1 && vx0) ? ly * (1.f - lx)         * msk : 0.f;
      float w11 = (vy1 && vx1) ? ly * lx                 * msk : 0.f;
      int y0c = min(max(y0, 0), HH - 1), y1c = min(max(y1, 0), HH - 1);
      int x0c = min(max(x0, 0), WW - 1), x1c = min(max(x1, 0), WW - 1);
      int i00 = y0c * WW + x0c, i01 = y0c * WW + x1c;
      int i10 = y1c * WW + x0c, i11 = y1c * WW + x1c;
      const __bf16* fub = fu + ((size_t)b * CC + g * 16 + c0) * HW_;
      v8bf sv;
#pragma unroll
      for (int i = 0; i < 8; ++i){
        const __bf16* fc = fub + i * HW_;
        float v = w00 * bf2f(fc[i00]) + w01 * bf2f(fc[i01])
                + w10 * bf2f(fc[i10]) + w11 * bf2f(fc[i11]);
        sv[i] = f2bf(v);
      }
      *(v8bf*)(lds + n * LDSD + pl * 16 + c0) = sv;
    }
    __syncthreads();
    v16bf bf0 = load_b(lds, 0, lane);
    v16bf bf1 = load_b(lds, 16, lane);
    v16bf a0  = load_a(Wd, wid * 32,      K2, kt * 32, lane);
    v16bf a1  = load_a(Wd, wid * 32 + 16, K2, kt * 32, lane);
    acc[0][0] = wmma_bf16(a0, bf0, acc[0][0]);
    acc[0][1] = wmma_bf16(a0, bf1, acc[0][1]);
    acc[1][0] = wmma_bf16(a1, bf0, acc[1][0]);
    acc[1][1] = wmma_bf16(a1, bf1, acc[1][1]);
    __syncthreads();
  }
  int col = lane & 15, hi = lane >> 4;
#pragma unroll
  for (int mt = 0; mt < 2; ++mt)
#pragma unroll
    for (int nt = 0; nt < 2; ++nt)
#pragma unroll
      for (int r = 0; r < 8; ++r){
        int oc = wid * 32 + mt * 16 + r + 8 * hi;
        int hw = hw0 + nt * 16 + col;
        float v = acc[mt][nt][r] + b_dcn[oc];
        v = fmaxf(v, 0.f) + feat_l[(b * CC + oc) * HW_ + hw];
        out[(b * CC + oc) * HW_ + hw] = v;
      }
}

extern "C" void kernel_launch(void* const* d_in, const int* in_sizes, int n_in,
                              void* d_out, int out_size, void* d_ws, size_t ws_size,
                              hipStream_t stream) {
  (void)in_sizes; (void)n_in; (void)out_size; (void)ws_size;
  const float* feat_l = (const float*)d_in[0];
  const float* feat_s = (const float*)d_in[1];
  const float* W_off  = (const float*)d_in[2];
  const float* W_om   = (const float*)d_in[3];
  const float* b_om   = (const float*)d_in[4];
  const float* W_dcn  = (const float*)d_in[5];
  const float* b_dcn  = (const float*)d_in[6];
  float* out = (float*)d_out;

  char* ws = (char*)d_ws;
  size_t off = 0;
  auto alloc = [&](size_t bytes) -> void* {
    void* p = ws + off;
    off += (bytes + 255) & ~(size_t)255;
    return p;
  };
  unsigned short* Wo    = (unsigned short*)alloc((size_t)CC * K1 * 2);       // 64 KB
  unsigned short* Wm    = (unsigned short*)alloc((size_t)OC2P * K2 * 2);     // 504 KB
  unsigned short* Wd    = (unsigned short*)alloc((size_t)CC * K2 * 2);       // 288 KB
  unsigned short* fu    = (unsigned short*)alloc((size_t)BB * CC * HW_ * 2); // 16 MB
  unsigned short* ofeat = (unsigned short*)alloc((size_t)BB * CC * HW_ * 2); // 16 MB
  float*          om    = (float*)alloc((size_t)BB * OC2 * HW_ * 4);         // 54 MB

  int nPixBlocks = (BB * HW_) / 32;  // 2048

  prep_weights<<<(OC2P * K2 + 255) / 256, 256, 0, stream>>>(W_off, W_om, W_dcn, Wo, Wm, Wd);
  upsample2x<<<(BB * CC * HW_) / 256, 256, 0, stream>>>(feat_s, fu);
  gemm1<<<nPixBlocks, 128, 0, stream>>>(feat_l, fu, Wo, ofeat);
  gemm2<<<nPixBlocks, 224, 0, stream>>>(ofeat, Wm, b_om, om);
  dcn_wmma<<<nPixBlocks, 128, 0, stream>>>(fu, om, Wd, b_dcn, feat_l, out);
}
